// MEFSSIM_65206193487944
// MI455X (gfx1250) — matrix operand order repaired and checked by reading, not compiled, not run on hardware
//
#include <hip/hip_runtime.h>
#include <hip/hip_bf16.h>
#include <math.h>

// ---------------- types ----------------
typedef __bf16 bhalf;
typedef __attribute__((ext_vector_type(16))) __bf16 v16bf;
typedef __attribute__((ext_vector_type(8)))  float  v8f;

// ---------------- geometry ----------------
#define H        1024
#define W        1024
#define CH       3
#define KIMG     4
#define WS       11
#define TP       5           // top pad
#define LP       7           // left pad (odd so bf16 pairs are 4B aligned)
#define PW       1056        // padded width  (>= LP + 1024 + 25)
#define PH       1040        // padded height (>= TP + 1024 + 10)
#define PLANE    (PW*PH)     // elements per padded plane
#define NPLANES  14          // 0:SX 1:SXX 2..5:SY 6..9:SYY 10..13:SXY
#define NTILES_X 64
#define NTILES   (64*64)

#define C1f      (0.01f*0.01f)
#define C2f      (0.03f*0.03f)
#define INV_DL   (1.0f/(2.0f*0.2f*0.2f))
#define INV_DG   (1.0f/(2.0f*0.2f*0.2f))

// ---------------- d_ws layout (bytes) ----------------
#define OFF_BFRAG    0            // 3*11*32*16 bf16 = 33792 B
#define OFF_LY       65536        // 4 f32
#define OFF_MUY_PART 65792        // 4096*4 f32 = 65536 B
#define OFF_TILESUM  131584       // 4096 f32
#define OFF_PLANES   262144       // 14 * PLANE bf16 (~30.8 MB)

// ---------------- fragment helpers (CDNA5 wave32 WMMA layouts) ----------------
// A (16-bit, 16x32): lane L holds row m=L&15; element e (r=e/2,p=e%2):
//   k = (r<4 ? 0:16) + (L<16 ? 0:8) + (r%4)*2 + p
// rowPtr points at (plane row y0+m+dy, column x0+2); hi = (lane&16)?8:0 is
// pre-added by the caller so all per-dy offsets below are compile-time
// immediates after unrolling (two contiguous 16B runs -> 2x global_load_b128).
__device__ inline v16bf load_afrag(const bhalf* rowPtr) {
  v16bf a;
#pragma unroll
  for (int r = 0; r < 8; ++r) {
    const int k = ((r < 4) ? 0 : 16) + (r & 3) * 2;
    a[2*r]     = rowPtr[k];
    a[2*r + 1] = rowPtr[k + 1];
  }
  return a;
}

// B (16-bit, 32x16): lane L holds column n=L&15; element e: k = (e<8?0:16)+(L<16?0:8)+(e&7)
// pre-packed contiguously: bt[(((f*11)+dy)*32 + lane)*16 + e]
__device__ inline v16bf load_bfrag(const bhalf* bt, int f, int dy, int lane) {
  const v16bf* p = (const v16bf*)(bt + (size_t)(((f * WS) + dy) * 32 + lane) * 16);
  return *p;
}

__device__ inline v8f wmma_bf16(v16bf a, v16bf b, v8f c) {
  return __builtin_amdgcn_wmma_f32_16x16x32_bf16(false, a, false, b, (short)0, c,
                                                 false, false);
}

__device__ inline float wave32_sum(float v) {
#pragma unroll
  for (int off = 16; off > 0; off >>= 1) v += __shfl_down(v, off, 32);
  return v;
}

// ---------------- K0: pack per-lane B fragments for the 3 filters ----------------
__global__ void k_weights(bhalf* bt) {
  float g[WS];
  {
    const float sigma = (float)WS / 6.0f;
    float s = 0.f;
#pragma unroll
    for (int i = 0; i < WS; ++i) { float d = (float)(i - WS/2); g[i] = __expf(-d*d/(2.f*sigma*sigma)); s += g[i]; }
#pragma unroll
    for (int i = 0; i < WS; ++i) g[i] /= s;
  }
  const int total = 3 * WS * 32 * 16;
  for (int idx = blockIdx.x * blockDim.x + threadIdx.x; idx < total;
       idx += gridDim.x * blockDim.x) {
    const int e    = idx & 15;
    const int lane = (idx >> 4) & 31;
    const int rem  = idx >> 9;
    const int dy   = rem % WS;
    const int f    = rem / WS;
    const int k    = ((lane & 16) ? 8 : 0) + ((e < 8) ? e : e + 8);
    const int n    = lane & 15;
    const int dx   = k - n;
    float val = 0.f;
    if (k < 26 && dx >= 0 && dx <= 10) {
      if (f == 0) {
        val = g[dy] * g[dx] * (1.0f / (float)CH);       // channel-avg gaussian
      } else {
        const float di = (float)(dy - WS/2), dj = (float)(dx - WS/2);
        const float den = di*di + dj*dj + 1e-6f;
        val = (f == 1) ? (dj / den) : (di / den);        // kx / ky (= kx^T)
      }
    }
    bt[idx] = (bhalf)val;
  }
}

// ---------------- K1: channel pre-sum into zero-padded bf16 planes ----------------
__global__ void k_presum(const float* __restrict__ X, const float* __restrict__ Ys,
                         bhalf* __restrict__ planes) {
  const int idx = blockIdx.x * blockDim.x + threadIdx.x;
  if (idx >= PLANE) return;
  const int py = idx / PW, px = idx % PW;
  const int y = py - TP, x = px - LP;
  float v[NPLANES];
  if ((unsigned)y < (unsigned)H && (unsigned)x < (unsigned)W) {
    const int HWs = H * W;
    float xc[CH], sx = 0.f, sxx = 0.f;
#pragma unroll
    for (int c = 0; c < CH; ++c) {
      xc[c] = X[c * HWs + y * W + x];
      sx += xc[c]; sxx += xc[c] * xc[c];
    }
    v[0] = sx; v[1] = sxx;
#pragma unroll
    for (int k = 0; k < KIMG; ++k) {
      float sy = 0.f, syy = 0.f, sxy = 0.f;
#pragma unroll
      for (int c = 0; c < CH; ++c) {
        const float yv = Ys[(k * CH + c) * HWs + y * W + x];
        sy += yv; syy += yv * yv; sxy += xc[c] * yv;
      }
      v[2 + k] = sy; v[6 + k] = syy; v[10 + k] = sxy;
    }
  } else {
#pragma unroll
    for (int p = 0; p < NPLANES; ++p) v[p] = 0.f;
  }
#pragma unroll
  for (int p = 0; p < NPLANES; ++p) planes[(size_t)p * PLANE + idx] = (bhalf)v[p];
}

// ---------------- K2a: gaussian conv of Y planes -> per-tile partial sums ----------------
__global__ void __launch_bounds__(32) k_muY(const bhalf* __restrict__ planes,
                                            const bhalf* __restrict__ bt,
                                            float* __restrict__ muy_part) {
  const int tile = blockIdx.x;
  const int x0 = (tile & (NTILES_X - 1)) * 16;
  const int y0 = (tile / NTILES_X) * 16;
  const int lane = threadIdx.x;
  const int m = lane & 15;
  const int hi = (lane & 16) ? 8 : 0;                   // A-layout K msb per half-wave
  const size_t laneOff = (size_t)(y0 + m) * PW + x0 + 2 + hi;
  for (int k = 0; k < KIMG; ++k) {
    const bhalf* base = planes + (size_t)(2 + k) * PLANE + laneOff;
    // burst-load the whole A column first so WMMAs retire on partial loadcnt
    v16bf a[WS];
#pragma unroll
    for (int dy = 0; dy < WS; ++dy) a[dy] = load_afrag(base + dy * PW);
    v8f acc;
#pragma unroll
    for (int i = 0; i < 8; ++i) acc[i] = 0.f;
#pragma unroll
    for (int dy = 0; dy < WS; ++dy)
      acc = wmma_bf16(a[dy], load_bfrag(bt, 0, dy, lane), acc);
    float s = 0.f;
#pragma unroll
    for (int r = 0; r < 8; ++r) s += acc[r];
    s = wave32_sum(s);
    if (lane == 0) muy_part[tile * KIMG + k] = s;
  }
}

__global__ void k_reduce_lY(const float* __restrict__ part, float* __restrict__ lY) {
  __shared__ float sh[256];
  const int t = threadIdx.x;
  for (int k = 0; k < KIMG; ++k) {
    float s = 0.f;
    for (int i = t; i < NTILES; i += blockDim.x) s += part[i * KIMG + k];
    sh[t] = s; __syncthreads();
    for (int off = blockDim.x / 2; off > 0; off >>= 1) {
      if (t < off) sh[t] += sh[t + off];
      __syncthreads();
    }
    if (t == 0) lY[k] = sh[0] * (1.0f / (float)(H * W));
    __syncthreads();
  }
}

// ---------------- K3: fused WMMA conv (14 planes x 3 filters) + SSIM math ----------------
__global__ void __launch_bounds__(32) k_mega(const bhalf* __restrict__ planes,
                                             const bhalf* __restrict__ bt,
                                             const float* __restrict__ lY,
                                             float* __restrict__ tilesum) {
  __shared__ float s_mu[NPLANES][256];
  __shared__ float s_ga[NPLANES][256];
  __shared__ float s_gn[NPLANES][256];

  const int tile = blockIdx.x;
  const int x0 = (tile & (NTILES_X - 1)) * 16;
  const int y0 = (tile / NTILES_X) * 16;
  const int lane = threadIdx.x;
  const int m = lane & 15;
  const int hi = (lane & 16) ? 8 : 0;
  const int mAdd = (lane & 16) ? 8 : 0;   // C/D: VGPR r, lanes16-31 hold M=r+8
  const int n = lane & 15;
  const size_t laneOff = (size_t)(y0 + m) * PW + x0 + 2 + hi;

  for (int p = 0; p < NPLANES; ++p) {
    const bhalf* base = planes + (size_t)p * PLANE + laneOff;
    if (p + 1 < NPLANES)  // pull next plane's tile toward the caches (global_prefetch_b8)
      __builtin_prefetch(planes + (size_t)(p + 1) * PLANE + laneOff, 0, 1);

    // burst-load the A column (22 b128 loads), then stream 33 WMMAs against it
    v16bf a[WS];
#pragma unroll
    for (int dy = 0; dy < WS; ++dy) a[dy] = load_afrag(base + dy * PW);

    v8f accG, accX, accY;
#pragma unroll
    for (int i = 0; i < 8; ++i) { accG[i] = 0.f; accX[i] = 0.f; accY[i] = 0.f; }
#pragma unroll
    for (int dy = 0; dy < WS; ++dy) {
      accG = wmma_bf16(a[dy], load_bfrag(bt, 0, dy, lane), accG);
      accX = wmma_bf16(a[dy], load_bfrag(bt, 1, dy, lane), accX);
      accY = wmma_bf16(a[dy], load_bfrag(bt, 2, dy, lane), accY);
    }
#pragma unroll
    for (int r = 0; r < 8; ++r) {
      const int pix = (r + mAdd) * 16 + n;
      const float gx = fminf(fmaxf(accX[r], -1.f), 1.f);
      const float gy = fminf(fmaxf(accY[r], -1.f), 1.f);
      const float gab = sqrtf(fmaxf(gx * gx + gy * gy, 1e-8f));
      const float gan = atanf(gy / fmaxf(gx, 1e-6f));
      s_mu[p][pix] = accG[r];
      s_ga[p][pix] = gab;
      s_gn[p][pix] = gan;
    }
  }
  __syncthreads();

  float lYl[KIMG];
#pragma unroll
  for (int k = 0; k < KIMG; ++k) lYl[k] = lY[k];

  float lsum = 0.f;
#pragma unroll
  for (int r = 0; r < 8; ++r) {
    const int pix = (r + mAdd) * 16 + n;
    const float muX = s_mu[0][pix];
    const float sX  = s_mu[1][pix] - muX * muX;

    float muYk[KIMG], sYk[KIMG];
    int kb = 0; float best = -3.4e38f;
#pragma unroll
    for (int k = 0; k < KIMG; ++k) {
      muYk[k] = s_mu[2 + k][pix];
      sYk[k]  = s_mu[6 + k][pix] - muYk[k] * muYk[k];
      if (sYk[k] > best) { best = sYk[k]; kb = k; }   // first-max == jnp.argmax
    }
    const float muY = muYk[kb], sY = sYk[kb];
    const float sXY = s_mu[10 + kb][pix] - muX * muY;
    const float cs  = (2.f * sXY + C2f) / fmaxf(sX + sY + C2f, 1e-6f);

    const float aX = s_ga[0][pix], nX = s_gn[0][pix];
    const float aY = s_ga[2 + kb][pix], nY = s_gn[2 + kb][pix];
    const float sAX  = fmaxf(s_ga[1][pix]      - aX * aX, 1e-6f);
    const float sAY  = fmaxf(s_ga[6 + kb][pix] - aY * aY, 1e-6f);
    const float sAXY = fmaxf(s_ga[10 + kb][pix] - aX * aY, 1e-6f);
    const float seqA = fminf(fmaxf(2.f * sAXY / fmaxf(sAX + sAY, 1e-6f), 0.f), 1.f);
    const float sNX  = fmaxf(s_gn[1][pix]      - nX * nX, 1e-6f);
    const float sNY  = fmaxf(s_gn[6 + kb][pix] - nY * nY, 1e-6f);
    const float sNXY = fmaxf(s_gn[10 + kb][pix] - nX * nY, 1e-6f);
    const float seqN = fminf(fmaxf(2.f * sNXY / fmaxf(sNX + sNY, 1e-6f), 0.f), 1.f);
    const float grad = seqA * seqN;

    float num = 0.f, den = 0.f;
#pragma unroll
    for (int k = 0; k < KIMG; ++k) {
      const float dL = muYk[k] - 0.5f;
      const float dG = lYl[k] - 0.5f;
      const float w = __expf(-dG * dG * INV_DG) * __expf(-dL * dL * INV_DL);
      num += w * muYk[k]; den += w;
    }
    const float muYw = num / den;
    const float l = (2.f * muX * muYw + C1f) / (muX * muX + muYw * muYw + C1f);

    lsum += l * cs * grad;
  }
  lsum = wave32_sum(lsum);
  if (lane == 0) tilesum[tile] = lsum;
}

// ---------------- K4: final mean ----------------
__global__ void k_final(const float* __restrict__ tilesum, float* __restrict__ out) {
  __shared__ float sh[256];
  const int t = threadIdx.x;
  float s = 0.f;
  for (int i = t; i < NTILES; i += blockDim.x) s += tilesum[i];
  sh[t] = s; __syncthreads();
  for (int off = blockDim.x / 2; off > 0; off >>= 1) {
    if (t < off) sh[t] += sh[t + off];
    __syncthreads();
  }
  if (t == 0) out[0] = sh[0] * (1.0f / (float)(H * W));
}

// ---------------- launch ----------------
extern "C" void kernel_launch(void* const* d_in, const int* in_sizes, int n_in,
                              void* d_out, int out_size, void* d_ws, size_t ws_size,
                              hipStream_t stream) {
  (void)in_sizes; (void)n_in; (void)out_size; (void)ws_size;
  const float* X  = (const float*)d_in[0];   // (1,3,1024,1024) f32
  const float* Ys = (const float*)d_in[1];   // (4,3,1024,1024) f32
  float* out = (float*)d_out;

  char* ws = (char*)d_ws;
  bhalf* bt      = (bhalf*)(ws + OFF_BFRAG);
  float* lY      = (float*)(ws + OFF_LY);
  float* muyp    = (float*)(ws + OFF_MUY_PART);
  float* tsum    = (float*)(ws + OFF_TILESUM);
  bhalf* planes  = (bhalf*)(ws + OFF_PLANES);

  k_weights<<<8, 256, 0, stream>>>(bt);
  k_presum<<<(PLANE + 255) / 256, 256, 0, stream>>>(X, Ys, planes);
  k_muY<<<NTILES, 32, 0, stream>>>(planes, bt, muyp);
  k_reduce_lY<<<1, 256, 0, stream>>>(muyp, lY);
  k_mega<<<NTILES, 32, 0, stream>>>(planes, bt, lY, tsum);
  k_final<<<1, 256, 0, stream>>>(tsum, out);
}
